// LSTMDecoder_73693048864830
// MI455X (gfx1250) — matrix operand order repaired and checked by reading, not compiled, not run on hardware
//
#include <hip/hip_runtime.h>

// ---------------------------------------------------------------------------
// LSTM decoder for MI455X (gfx1250, wave32, WMMA).
// - Weights converted once to bf16 (~54MB) -> resident in 192MB L2.
// - All GEMMs via v_wmma_f32_16x16x32_bf16; logits/wv GEMMs use 2x2 register blocking so each
//   WMMA costs ~2 b128 loads (VMEM-issue headroom).
// - Deterministic split-K (no float atomics) for the softmax@embedding GEMM.
// ---------------------------------------------------------------------------

typedef __attribute__((ext_vector_type(16))) __bf16 bf16x16;
typedef __attribute__((ext_vector_type(8)))  float  f32x8;
typedef __attribute__((ext_vector_type(4)))  unsigned int uintx4;

union FragBF {            // 8 VGPRs = 16 bf16 per lane
    bf16x16 bf;
    uintx4  u4[2];
};

#define DEVI static __device__ __forceinline__

DEVI unsigned short f2bf(float f) {            // round-to-nearest-even f32->bf16
    unsigned int u = __float_as_uint(f);
    u += 0x7FFFu + ((u >> 16) & 1u);
    return (unsigned short)(u >> 16);
}

DEVI f32x8 wmma_bf16(const FragBF& a, const FragBF& b, f32x8 c) {
    return __builtin_amdgcn_wmma_f32_16x16x32_bf16(
        false, a.bf, false, b.bf, (short)0, c, false, false);
}

// Problem sizes
#define BB 32
#define TT 64
#define DWV 256
#define DIN 256
#define HH 512
#define VV 32000
#define KSPLIT 8            // split-K chunks for wv GEMM (32000 = 8*125*32)

// ---------------------------------------------------------------------------
// One-time (per launch) weight conversion kernels
// ---------------------------------------------------------------------------

// W_cat[n, 0:512]=W_ih[n,:], W_cat[n,512:1024]=W_hh[n,:]   (bf16, row-major)
__global__ void conv_wcat(const float* __restrict__ W_ih, const float* __restrict__ W_hh,
                          unsigned short* __restrict__ Wcat) {
    int idx = blockIdx.x * 256 + threadIdx.x;      // 2048*1024
    int n = idx >> 10, k = idx & 1023;
    float v = (k < 512) ? W_ih[n * 512 + k] : W_hh[n * 512 + (k - 512)];
    Wcat[idx] = f2bf(v);
}

// straight f32 -> bf16 copy (W_fc, row-major (V,512) preserved)
__global__ void conv_wfc(const float* __restrict__ W, unsigned short* __restrict__ Wb) {
    int idx = blockIdx.x * 256 + threadIdx.x;      // 32000*512
    Wb[idx] = f2bf(W[idx]);
}

// Embedding (V=32000, 256) -> WMMA B-fragment friendly packing:
// dword[(kg*256 + n)*8 + j] = { bf16(E[kg*16+2j][n]), bf16(E[kg*16+2j+1][n]) }
// so each lane's 8 B-frag VGPRs are two contiguous b128 loads.
__global__ void conv_epack(const float* __restrict__ E, unsigned int* __restrict__ Ep) {
    int idx = blockIdx.x * 256 + threadIdx.x;      // 2000*256*8 = 4,096,000
    int j  = idx & 7;
    int n  = (idx >> 3) & 255;
    int kg = idx >> 11;
    int k0 = kg * 16 + 2 * j;
    unsigned int lo = f2bf(E[(k0    ) * DWV + n]);
    unsigned int hi = f2bf(E[(k0 + 1) * DWV + n]);
    Ep[idx] = lo | (hi << 16);
}

// h=0, c=0, wv = input_seq_emb[:,0,:]
__global__ void init_state(const float* __restrict__ ise,
                           unsigned short* __restrict__ hbf, float* __restrict__ cst,
                           float* __restrict__ wv) {
    int idx = blockIdx.x * 256 + threadIdx.x;      // 16384
    hbf[idx] = 0;
    cst[idx] = 0.f;
    if (idx < BB * DWV) {
        int b = idx >> 8, k = idx & 255;
        wv[idx] = ise[b * (TT * DWV) + k];         // t = 0 slice
    }
}

// ---------------------------------------------------------------------------
// Per-timestep kernels
// ---------------------------------------------------------------------------

// xh[b, 0:256]=bf16(wv) | [256:512]=bf16(emb[b,t]) | [512:1024]=h_{t-1} (bf16)
__global__ void prep_x(const float* __restrict__ wv, const float* __restrict__ emb,
                       const unsigned short* __restrict__ hbf,
                       unsigned short* __restrict__ xh, int t) {
    int idx = blockIdx.x * 256 + threadIdx.x;      // 32*1024
    int b = idx >> 10, k = idx & 1023;
    unsigned short v;
    if (k < 256)      v = f2bf(wv[b * DWV + k]);
    else if (k < 512) v = f2bf(emb[(b * TT + t) * DIN + (k - 256)]);
    else              v = hbf[b * HH + (k - 512)];
    xh[idx] = v;
}

// gates_pre = xh(32x1024) @ Wcat^T(1024x2048).
// One wave per n-tile computes both m-tiles (B fragments reused 2x).
__global__ void gates_gemm(const unsigned short* __restrict__ xh,
                           const unsigned short* __restrict__ Wcat,
                           float* __restrict__ gates) {
    int lane = threadIdx.x & 31;
    int wave = threadIdx.x >> 5;
    int nTile = blockIdx.x * 8 + wave;             // 0..127
    int half = lane >> 4, lid = lane & 15;
    const unsigned short* a0 = xh + lid * 1024 + half * 8;
    const unsigned short* a1 = a0 + 16 * 1024;
    int nBase = nTile * 16;
    const unsigned short* bRow = Wcat + (nBase + lid) * 1024 + half * 8;
    f32x8 acc0 = {}, acc1 = {};
    for (int kk = 0; kk < 32; ++kk) {              // K = 1024
        FragBF A0, A1, B;
        A0.u4[0] = *(const uintx4*)(a0 + kk * 32);
        A0.u4[1] = *(const uintx4*)(a0 + kk * 32 + 16);
        A1.u4[0] = *(const uintx4*)(a1 + kk * 32);
        A1.u4[1] = *(const uintx4*)(a1 + kk * 32 + 16);
        B.u4[0]  = *(const uintx4*)(bRow + kk * 32);
        B.u4[1]  = *(const uintx4*)(bRow + kk * 32 + 16);
        acc0 = wmma_bf16(A0, B, acc0);
        acc1 = wmma_bf16(A1, B, acc1);
    }
    int row = half * 8;
    int col = nBase + lid;
#pragma unroll
    for (int r = 0; r < 8; ++r) {
        gates[(row + r)      * 2048 + col] = acc0[r];
        gates[(row + 16 + r) * 2048 + col] = acc1[r];
    }
}

// LSTM cell elementwise: c,h update; h also stored bf16 for next GEMMs.
__global__ void cell_update(const float* __restrict__ gates,
                            const float* __restrict__ b_ih, const float* __restrict__ b_hh,
                            float* __restrict__ cst, unsigned short* __restrict__ hbf,
                            float* __restrict__ outH, float* __restrict__ outC, int last) {
    int idx = blockIdx.x * 256 + threadIdx.x;      // 32*512
    int b = idx >> 9, j = idx & 511;
    const float* g = gates + b * 2048;
    float gi = g[j]         + b_ih[j]          + b_hh[j];
    float gf = g[512 + j]   + b_ih[512 + j]    + b_hh[512 + j];
    float gg = g[1024 + j]  + b_ih[1024 + j]   + b_hh[1024 + j];
    float go = g[1536 + j]  + b_ih[1536 + j]   + b_hh[1536 + j];
    float si = 1.f / (1.f + __expf(-gi));
    float sf = 1.f / (1.f + __expf(-gf));
    float tg = tanhf(gg);
    float so = 1.f / (1.f + __expf(-go));
    float cN = sf * cst[idx] + si * tg;
    float hN = so * tanhf(cN);
    cst[idx] = cN;
    hbf[idx] = f2bf(hN);
    if (last) { outH[idx] = hN; outC[idx] = cN; }
}

// logits = h(32x512) @ W_fc^T(512x32000) + b_fc, written to d_out[b,t,:].
// 2 M-tiles x 2 N-tiles per wave: 4 WMMA per 8 b128 loads (2 loads/WMMA).
__global__ void logits_gemm(const unsigned short* __restrict__ hbf,
                            const unsigned short* __restrict__ Wfc,
                            const float* __restrict__ b_fc,
                            float* __restrict__ out, int t) {
    int lane = threadIdx.x & 31;
    int wave = threadIdx.x >> 5;
    int p = blockIdx.x * 8 + wave;                 // 0..999 n-tile pairs
    int half = lane >> 4, lid = lane & 15;
    const unsigned short* a0 = hbf + lid * HH + half * 16;
    const unsigned short* a1 = a0 + 16 * HH;
    int n0 = p * 32 + lid;
    int n1 = n0 + 16;
    const unsigned short* bR0 = Wfc + (size_t)n0 * HH + half * 16;
    const unsigned short* bR1 = Wfc + (size_t)n1 * HH + half * 16;
    f32x8 acc00 = {}, acc10 = {}, acc01 = {}, acc11 = {};
    for (int kk = 0; kk < 16; ++kk) {              // K = 512
        FragBF A0, A1, B0, B1;
        A0.u4[0] = *(const uintx4*)(a0 + kk * 32);
        A0.u4[1] = *(const uintx4*)(a0 + kk * 32 + 8);
        A1.u4[0] = *(const uintx4*)(a1 + kk * 32);
        A1.u4[1] = *(const uintx4*)(a1 + kk * 32 + 8);
        B0.u4[0] = *(const uintx4*)(bR0 + kk * 32);
        B0.u4[1] = *(const uintx4*)(bR0 + kk * 32 + 8);
        B1.u4[0] = *(const uintx4*)(bR1 + kk * 32);
        B1.u4[1] = *(const uintx4*)(bR1 + kk * 32 + 8);
        if (kk + 2 < 16) {
            __builtin_prefetch(bR0 + (kk + 2) * 32, 0, 0);
            __builtin_prefetch(bR1 + (kk + 2) * 32, 0, 0);
        }
        acc00 = wmma_bf16(A0, B0, acc00);
        acc10 = wmma_bf16(A1, B0, acc10);
        acc01 = wmma_bf16(A0, B1, acc01);
        acc11 = wmma_bf16(A1, B1, acc11);
    }
    float bias0 = b_fc[n0], bias1 = b_fc[n1];
    int m0 = half * 8;
#pragma unroll
    for (int r = 0; r < 8; ++r) {
        out[((size_t)(m0 + r)      * TT + t) * VV + n0] = acc00[r] + bias0;
        out[((size_t)(m0 + 16 + r) * TT + t) * VV + n0] = acc10[r] + bias0;
        out[((size_t)(m0 + r)      * TT + t) * VV + n1] = acc01[r] + bias1;
        out[((size_t)(m0 + 16 + r) * TT + t) * VV + n1] = acc11[r] + bias1;
    }
}

// Per-row softmax statistics (max, 1/sumexp).
__global__ void softmax_stats(const float* __restrict__ out, int t,
                              float* __restrict__ rowmax, float* __restrict__ rowrsum) {
    int b = blockIdx.x;
    const float* row = out + (size_t)(b * TT + t) * VV;
    __shared__ float red[256];
    float m = -1e30f;
    for (int v = threadIdx.x; v < VV; v += 256) m = fmaxf(m, row[v]);
    red[threadIdx.x] = m; __syncthreads();
    for (int s = 128; s > 0; s >>= 1) {
        if (threadIdx.x < s) red[threadIdx.x] = fmaxf(red[threadIdx.x], red[threadIdx.x + s]);
        __syncthreads();
    }
    m = red[0]; __syncthreads();
    float sum = 0.f;
    for (int v = threadIdx.x; v < VV; v += 256) sum += __expf(row[v] - m);
    red[threadIdx.x] = sum; __syncthreads();
    for (int s = 128; s > 0; s >>= 1) {
        if (threadIdx.x < s) red[threadIdx.x] += red[threadIdx.x + s];
        __syncthreads();
    }
    if (threadIdx.x == 0) { rowmax[b] = m; rowrsum[b] = 1.f / red[0]; }
}

// P = softmax probabilities in bf16 (A-operand of the wv GEMM).
__global__ void softmax_p(const float* __restrict__ out, int t,
                          const float* __restrict__ rowmax, const float* __restrict__ rowrsum,
                          unsigned short* __restrict__ P) {
    int idx = blockIdx.x * 256 + threadIdx.x;      // 32*32000
    int b = idx / VV, v = idx % VV;
    float p = __expf(out[(size_t)(b * TT + t) * VV + v] - rowmax[b]) * rowrsum[b];
    P[idx] = f2bf(p);
}

// wv partials = P(32x32000) @ E(32000x256), deterministic split-K.
// Each wave: both M-tiles x one N-tile-pair x one K-chunk (4 accumulators).
__global__ void wv_gemm(const unsigned short* __restrict__ P,
                        const unsigned int* __restrict__ Epack,
                        float* __restrict__ part) {
    int lane = threadIdx.x & 31;
    int wave = threadIdx.x >> 5;
    int id = blockIdx.x * 8 + wave;                // 0..63
    int kchunk = id >> 3;                          // 0..7
    int npair  = id & 7;                           // 0..7 (256/32)
    int half = lane >> 4, lid = lane & 15;
    const unsigned short* a0 = P + lid * VV + half * 16;
    const unsigned short* a1 = a0 + 16 * VV;
    int n0 = npair * 32 + lid;
    int n1 = n0 + 16;
    f32x8 acc00 = {}, acc10 = {}, acc01 = {}, acc11 = {};
    int kk0 = kchunk * 125;
    for (int kk = kk0; kk < kk0 + 125; ++kk) {     // total K = 1000*32 = 32000
        FragBF A0, A1, B0, B1;
        A0.u4[0] = *(const uintx4*)(a0 + kk * 32);
        A0.u4[1] = *(const uintx4*)(a0 + kk * 32 + 8);
        A1.u4[0] = *(const uintx4*)(a1 + kk * 32);
        A1.u4[1] = *(const uintx4*)(a1 + kk * 32 + 8);
        const unsigned int* bp0 = Epack + (((kk * 2 + half) * 256) + n0) * 8;
        const unsigned int* bp1 = Epack + (((kk * 2 + half) * 256) + n1) * 8;
        B0.u4[0] = *(const uintx4*)(bp0);
        B0.u4[1] = *(const uintx4*)(bp0 + 4);
        B1.u4[0] = *(const uintx4*)(bp1);
        B1.u4[1] = *(const uintx4*)(bp1 + 4);
        acc00 = wmma_bf16(A0, B0, acc00);
        acc10 = wmma_bf16(A1, B0, acc10);
        acc01 = wmma_bf16(A0, B1, acc01);
        acc11 = wmma_bf16(A1, B1, acc11);
    }
    float* pc = part + kchunk * (BB * DWV);
    int m0 = half * 8;
#pragma unroll
    for (int r = 0; r < 8; ++r) {
        pc[(m0 + r)      * DWV + n0] = acc00[r];
        pc[(m0 + 16 + r) * DWV + n0] = acc10[r];
        pc[(m0 + r)      * DWV + n1] = acc01[r];
        pc[(m0 + 16 + r) * DWV + n1] = acc11[r];
    }
}

// Fixed-order split-K reduction (bit-deterministic across replays).
__global__ void wv_reduce(const float* __restrict__ part, float* __restrict__ wv) {
    int idx = blockIdx.x * 256 + threadIdx.x;      // 32*256
    float s = 0.f;
#pragma unroll
    for (int c = 0; c < KSPLIT; ++c) s += part[c * (BB * DWV) + idx];
    wv[idx] = s;
}

// ---------------------------------------------------------------------------
// Host launcher
// ---------------------------------------------------------------------------
extern "C" void kernel_launch(void* const* d_in, const int* in_sizes, int n_in,
                              void* d_out, int out_size, void* d_ws, size_t ws_size,
                              hipStream_t stream) {
    const float* emb  = (const float*)d_in[0];
    const float* ise  = (const float*)d_in[1];
    const float* E    = (const float*)d_in[2];
    const float* W_ih = (const float*)d_in[3];
    const float* W_hh = (const float*)d_in[4];
    const float* b_ih = (const float*)d_in[5];
    const float* b_hh = (const float*)d_in[6];
    const float* W_fc = (const float*)d_in[7];
    const float* b_fc = (const float*)d_in[8];

    float* out  = (float*)d_out;
    float* outH = out + (size_t)BB * TT * VV;
    float* outC = outH + BB * HH;

    // Workspace carve-up (~56 MB total)
    char* w = (char*)d_ws;
    size_t need = 0;
#define CARVE(ptr, type, count) type* ptr = (type*)(w + need); need += (size_t)(count) * sizeof(type);
    CARVE(Wcat,  unsigned short, 2048 * 1024)          // bf16 [W_ih|W_hh]
    CARVE(Wfc16, unsigned short, (size_t)VV * HH)      // bf16 W_fc
    CARVE(Epack, unsigned int,   (size_t)(VV / 16) * 256 * 8)
    CARVE(xh,    unsigned short, BB * 1024)
    CARVE(hbf,   unsigned short, BB * HH)
    CARVE(cst,   float,          BB * HH)
    CARVE(gates, float,          BB * 2048)
    CARVE(wv,    float,          BB * DWV)
    CARVE(part,  float,          KSPLIT * BB * DWV)
    CARVE(P,     unsigned short, (size_t)BB * VV)
    CARVE(rowmax,  float, 64)
    CARVE(rowrsum, float, 64)
#undef CARVE
    if (ws_size < need) return;                        // insufficient scratch

    // One-time conversions (cheap streaming passes; weights then live in L2)
    conv_wcat <<<(2048 * 1024) / 256, 256, 0, stream>>>(W_ih, W_hh, Wcat);
    conv_wfc  <<<((size_t)VV * HH) / 256, 256, 0, stream>>>(W_fc, Wfc16);
    conv_epack<<<((VV / 16) * 256 * 8) / 256, 256, 0, stream>>>(E, Epack);
    init_state<<<(BB * HH) / 256, 256, 0, stream>>>(ise, hbf, cst, wv);

    for (int t = 0; t < TT; ++t) {
        prep_x     <<<(BB * 1024) / 256, 256, 0, stream>>>(wv, emb, hbf, xh, t);
        gates_gemm <<<16,  256, 0, stream>>>(xh, Wcat, gates);
        cell_update<<<(BB * HH) / 256, 256, 0, stream>>>(gates, b_ih, b_hh, cst, hbf,
                                                         outH, outC, t == TT - 1);
        logits_gemm<<<125, 256, 0, stream>>>(hbf, Wfc16, b_fc, out, t);
        softmax_stats<<<BB, 256, 0, stream>>>(out, t, rowmax, rowrsum);
        softmax_p  <<<(BB * VV) / 256, 256, 0, stream>>>(out, t, rowmax, rowrsum, P);
        wv_gemm    <<<KSPLIT, 256, 0, stream>>>(P, Epack, part);
        wv_reduce  <<<(BB * DWV) / 256, 256, 0, stream>>>(part, wv);
    }
}
